// HeteroEncoder_38740605009975
// MI455X (gfx1250) — matrix operand order repaired and checked by reading, not compiled, not run on hardware
//
#include <hip/hip_runtime.h>

// ============================================================================
// HGT HeteroEncoder for MI455X (gfx1250, wave32, WMMA).
//
// Strategy (memory-bound workload, 23.3 TB/s HBM, 192MB L2):
//  - Rewrite per-edge relation transforms as per-source-node GEMMs:
//      k_rel[ei][n] = K[n] @ A_rel[ei]  (per head 32x32), same for V/M_rel.
//    Per-edge work becomes a gather + 128-wide dot + scatter softmax.
//  - All GEMMs via v_wmma_f32_16x16x32_bf16 (f32 accumulate, bf16 operands),
//    staged through LDS with conflict-free 17-dword pitch.
//  - Scatter softmax: atomic-max (order-preserving uint map) + atomic f32 adds.
//  - Epilogues: block-uniform full-tile fast path (no per-element exec juggling).
// ============================================================================

typedef __attribute__((ext_vector_type(16))) __bf16 v16bf;
typedef __attribute__((ext_vector_type(8)))  float  v8f;

#define AP 34  // LDS pitch in bf16 elements (17 dwords -> conflict-free)

static __device__ inline __bf16 f2bf(float f) {
  unsigned u = __float_as_uint(f);
  u += 0x7fffu + ((u >> 16) & 1u);          // round-to-nearest-even
  unsigned short s = (unsigned short)(u >> 16);
  return __builtin_bit_cast(__bf16, s);
}

// Order-preserving float<->uint map for atomic max on signed floats.
static __device__ inline unsigned mapf(float f) {
  unsigned u = __float_as_uint(f);
  return (u & 0x80000000u) ? ~u : (u | 0x80000000u);
}
static __device__ inline float unmapf(unsigned u) {
  return __uint_as_float((u & 0x80000000u) ? (u & 0x7fffffffu) : ~u);
}

// --- WMMA fragment loads (contiguous k = 16*(lane/16) + s, same in A and B) ------------------
// A 16x32: lane<16 -> M=lane,  K=0..15
//          lane>=16 -> M=lane-16, K=16..31
static __device__ inline v16bf load_a_frag(const __bf16* As, int tr, int lane) {
  const int m   = lane & 15;
  const int kb1 = (lane < 16) ? 0 : 16;
  const __bf16* p = As + (tr + m) * AP;
  v16bf a;
#pragma unroll
  for (int j = 0; j < 8; ++j) a[j]     = p[kb1 + j];
#pragma unroll
  for (int j = 0; j < 8; ++j) a[8 + j] = p[kb1 + 8 + j];
  return a;
}
// B 32x16: lane<16 -> N=lane, K=0..15 ; lane>=16 -> N=lane-16, K=16..31
// Bs stored transposed: Bs[col*AP + k]
static __device__ inline v16bf load_b_frag(const __bf16* Bs, int tc, int lane) {
  const int n  = lane & 15;
  const int kb = (lane < 16) ? 0 : 16;
  const __bf16* p = Bs + (tc + n) * AP + kb;
  v16bf b;
#pragma unroll
  for (int j = 0; j < 16; ++j) b[j] = p[j];
  return b;
}

enum { E_NONE = 0, E_RELU = 1, E_OUT = 2 };

// ============================================================================
// Generic GEMM: out[N,M] = epilogue(A[N,K] @ W[K,M] + bias)
// Block: 256 threads (8 waves), tile 32 rows x 64 cols, K step 32.
// E_OUT: out = relu(sig(skip)*h + (1-sig(skip))*xprev), h = acc + bias.
// ============================================================================
template <int EPI>
__global__ void __launch_bounds__(256)
hgt_gemm(const float* __restrict__ A, const float* __restrict__ Wt,
         const float* __restrict__ bias, float* __restrict__ out,
         int N, int K, int M,
         const float* __restrict__ xprev, const float* __restrict__ skip) {
  __shared__ __bf16 As[32 * AP];
  __shared__ __bf16 Bs[64 * AP];
  const int tid  = threadIdx.x;
  const int lane = tid & 31;
  const int wave = tid >> 5;
  const int rowBase = blockIdx.x * 32;
  const int colBase = blockIdx.y * 64;
  const int tr = (wave >> 2) * 16;   // 0 or 16 within 32 rows
  const int tc = (wave & 3) * 16;    // 0..48 within 64 cols

  // per-thread staging coordinates
  const int lr  = tid >> 3;          // A tile row     (0..31)
  const int lk  = (tid & 7) * 4;     // A tile k       (0..28)
  const int wk  = tid >> 4;          // W tile k       (0..15, x2 passes)
  const int wc  = (tid & 15) * 4;    // W tile col     (0..60)

  v8f acc = {};
  for (int k0 = 0; k0 < K; k0 += 32) {
    { // A tile 32x32 (f32 -> bf16)
      int gr = rowBase + lr;
      float4 v = make_float4(0.f, 0.f, 0.f, 0.f);
      if (gr < N) {
        const float* pa = A + (size_t)gr * K + k0 + lk;
        v = *(const float4*)pa;
        if (k0 + 32 < K) __builtin_prefetch(pa + 32, 0, 3);  // next K tile
      }
      __bf16* p = &As[lr * AP + lk];
      p[0] = f2bf(v.x); p[1] = f2bf(v.y); p[2] = f2bf(v.z); p[3] = f2bf(v.w);
    }
    { // W tile 32x64, stored transposed Bs[col][k]
#pragma unroll
      for (int pass = 0; pass < 2; ++pass) {
        int kr = wk + pass * 16;
        const float* pw = Wt + (size_t)(k0 + kr) * M + colBase + wc;
        const float4 v = *(const float4*)pw;
        if (k0 + 32 < K) __builtin_prefetch(pw + (size_t)32 * M, 0, 3);
        Bs[(wc + 0) * AP + kr] = f2bf(v.x);
        Bs[(wc + 1) * AP + kr] = f2bf(v.y);
        Bs[(wc + 2) * AP + kr] = f2bf(v.z);
        Bs[(wc + 3) * AP + kr] = f2bf(v.w);
      }
    }
    __syncthreads();
    v16bf av = load_a_frag(As, tr, lane);
    v16bf bv = load_b_frag(Bs, tc, lane);
    acc = __builtin_amdgcn_wmma_f32_16x16x32_bf16(false, av, false, bv,
                                                  (short)0, acc, false, false);
    __syncthreads();
  }

  // ---- epilogue: C/D layout VGPR r -> M = r (+8 for hi lanes), N = lane&15
  const int cn   = colBase + tc + (lane & 15);
  const int rAdd = (lane < 16) ? 0 : 8;
  const int row0 = rowBase + tr + rAdd;
  const float bv0 = bias[cn];
  float g = 0.f;
  if (EPI == E_OUT) g = 1.f / (1.f + expf(-skip[0]));

  float* po = out + (size_t)row0 * M + cn;
  const float* xp = (EPI == E_OUT) ? xprev + (size_t)row0 * 128 + cn : nullptr;

  if (rowBase + 32 <= N) {  // block-uniform full-tile fast path
#pragma unroll
    for (int r = 0; r < 8; ++r) {
      float h = acc[r] + bv0;
      float o;
      if (EPI == E_RELU)     o = fmaxf(h, 0.f);
      else if (EPI == E_OUT) o = fmaxf(g * h + (1.f - g) * xp[(size_t)r * 128], 0.f);
      else                   o = h;
      po[(size_t)r * M] = o;
    }
  } else {
#pragma unroll
    for (int r = 0; r < 8; ++r) {
      if (row0 + r < N) {
        float h = acc[r] + bv0;
        float o;
        if (EPI == E_RELU)     o = fmaxf(h, 0.f);
        else if (EPI == E_OUT) o = fmaxf(g * h + (1.f - g) * xp[(size_t)r * 128], 0.f);
        else                   o = h;
        po[(size_t)r * M] = o;
      }
    }
  }
}

// ============================================================================
// Per-head relation transform: Y[n, h*32+f] = sum_d X[n*384 + h*32 + d] * R[h,d,f]
// X is a K- or V-slice of the fused kqv buffer (row stride 384). K=32 -> 1 WMMA.
// Block: 128 threads (4 waves), tile 32 rows x 32 cols; grid.y = head.
// ============================================================================
__global__ void __launch_bounds__(128)
hgt_rel(const float* __restrict__ X, const float* __restrict__ R,
        float* __restrict__ Y, int N) {
  __shared__ __bf16 As[32 * AP];
  __shared__ __bf16 Bs[32 * AP];
  const int tid  = threadIdx.x;
  const int lane = tid & 31;
  const int wave = tid >> 5;
  const int rowBase = blockIdx.x * 32;
  const int h = blockIdx.y;

  { // X tile 32 rows x 32 d
    int r  = tid >> 2;
    int kk = (tid & 3) * 8;
    int gr = rowBase + r;
    float4 a = make_float4(0.f, 0.f, 0.f, 0.f), b = a;
    if (gr < N) {
      const float* p = X + (size_t)gr * 384 + h * 32 + kk;
      a = *(const float4*)p;
      b = *(const float4*)(p + 4);
    }
    __bf16* q = &As[r * AP + kk];
    q[0] = f2bf(a.x); q[1] = f2bf(a.y); q[2] = f2bf(a.z); q[3] = f2bf(a.w);
    q[4] = f2bf(b.x); q[5] = f2bf(b.y); q[6] = f2bf(b.z); q[7] = f2bf(b.w);
  }
  { // R[h] 32x32, store transposed Bs[f][d]
    int d = tid >> 2;
    int f = (tid & 3) * 8;
    const float4 a = *(const float4*)(R + h * 1024 + d * 32 + f);
    const float4 b = *(const float4*)(R + h * 1024 + d * 32 + f + 4);
    Bs[(f + 0) * AP + d] = f2bf(a.x); Bs[(f + 1) * AP + d] = f2bf(a.y);
    Bs[(f + 2) * AP + d] = f2bf(a.z); Bs[(f + 3) * AP + d] = f2bf(a.w);
    Bs[(f + 4) * AP + d] = f2bf(b.x); Bs[(f + 5) * AP + d] = f2bf(b.y);
    Bs[(f + 6) * AP + d] = f2bf(b.z); Bs[(f + 7) * AP + d] = f2bf(b.w);
  }
  __syncthreads();

  const int tr = (wave >> 1) * 16;
  const int tc = (wave & 1) * 16;
  v16bf av = load_a_frag(As, tr, lane);
  v16bf bv = load_b_frag(Bs, tc, lane);
  v8f acc = {};
  acc = __builtin_amdgcn_wmma_f32_16x16x32_bf16(false, av, false, bv,
                                                (short)0, acc, false, false);

  const int cn   = h * 32 + tc + (lane & 15);
  const int rAdd = (lane < 16) ? 0 : 8;
  const int row0 = rowBase + tr + rAdd;
  float* po = Y + (size_t)row0 * 128 + cn;
  if (rowBase + 32 <= N) {  // full-tile fast path
#pragma unroll
    for (int r = 0; r < 8; ++r) po[(size_t)r * 128] = acc[r];
  } else {
#pragma unroll
    for (int r = 0; r < 8; ++r)
      if (row0 + r < N) po[(size_t)r * 128] = acc[r];
  }
}

// ============================================================================
// Edge pass 1: a[e,h] = dot(krel[src], q[dst]) * p_rel[h] / sqrt(32);
//              atomic-max into amax. One wave per edge; lane -> 4 features.
// ============================================================================
__global__ void __launch_bounds__(256)
hgt_pass1(const float* __restrict__ krel, const float* __restrict__ q,
          const int* __restrict__ src, const int* __restrict__ dst, int E,
          const float* __restrict__ prel, float* __restrict__ abuf,
          unsigned* __restrict__ amax, int dstOff) {
  const int lane = threadIdx.x & 31;
  const int nw = gridDim.x * (blockDim.x >> 5);
  for (int e = blockIdx.x * (blockDim.x >> 5) + (threadIdx.x >> 5); e < E; e += nw) {
    int s = src[e], t = dst[e];
    const float4 kv = *(const float4*)(krel + (size_t)s * 128 + lane * 4);
    const float4 qv = *(const float4*)(q    + (size_t)t * 384 + lane * 4);
    float p = kv.x * qv.x + kv.y * qv.y + kv.z * qv.z + kv.w * qv.w;
    p += __shfl_xor(p, 1, 32);   // reduce within 8-lane head group
    p += __shfl_xor(p, 2, 32);
    p += __shfl_xor(p, 4, 32);
    if ((lane & 7) == 0) {
      int h = lane >> 3;
      float a = p * prel[h] * 0.17677669529663687f;  // 1/sqrt(32)
      abuf[(size_t)e * 4 + h] = a;
      atomicMax(&amax[(size_t)(dstOff + t) * 4 + h], mapf(a));
    }
  }
}

// Edge pass 2: ex = exp(a - amax[dst,h]); abuf=ex; atomicAdd den.
__global__ void __launch_bounds__(256)
hgt_pass2(float* __restrict__ abuf, const int* __restrict__ dst, int E4,
          const unsigned* __restrict__ amax, float* __restrict__ den, int dstOff) {
  for (int i = blockIdx.x * blockDim.x + threadIdx.x; i < E4;
       i += gridDim.x * blockDim.x) {
    int e = i >> 2, h = i & 3;
    int t = dst[e];
    float am = unmapf(amax[(size_t)(dstOff + t) * 4 + h]);
    float ex = expf(abuf[i] - am);
    abuf[i] = ex;
    atomicAdd(&den[(size_t)(dstOff + t) * 4 + h], ex);
  }
}

// Edge pass 3: w = ex/(den+1e-16); agg[dst] += w * vrel[src]. Wave per edge.
__global__ void __launch_bounds__(256)
hgt_pass3(const float* __restrict__ abuf, const float* __restrict__ vrel,
          const int* __restrict__ src, const int* __restrict__ dst, int E,
          const float* __restrict__ den, float* __restrict__ agg, int dstOff) {
  const int lane = threadIdx.x & 31;
  const int nw = gridDim.x * (blockDim.x >> 5);
  for (int e = blockIdx.x * (blockDim.x >> 5) + (threadIdx.x >> 5); e < E; e += nw) {
    int s = src[e], t = dst[e];
    int h = lane >> 3;
    float w = abuf[(size_t)e * 4 + h] /
              (den[(size_t)(dstOff + t) * 4 + h] + 1e-16f);
    const float4 v = *(const float4*)(vrel + (size_t)s * 128 + lane * 4);
    float* o = agg + (size_t)(dstOff + t) * 128 + lane * 4;
    atomicAdd(o + 0, w * v.x);
    atomicAdd(o + 1, w * v.y);
    atomicAdd(o + 2, w * v.z);
    atomicAdd(o + 3, w * v.w);
  }
}

__global__ void hgt_fill_u32(unsigned* p, unsigned v, size_t n) {
  size_t i = (size_t)blockIdx.x * blockDim.x + threadIdx.x;
  if (i < n) p[i] = v;
}
__global__ void hgt_fill_f32(float* p, float v, size_t n) {
  size_t i = (size_t)blockIdx.x * blockDim.x + threadIdx.x;
  if (i < n) p[i] = v;
}
__global__ void hgt_gelu(float* p, size_t n) {  // exact gelu
  size_t i = (size_t)blockIdx.x * blockDim.x + threadIdx.x;
  if (i < n) {
    float x = p[i];
    p[i] = 0.5f * x * (1.f + erff(x * 0.70710678118654752f));
  }
}

// ============================================================================
// Host orchestration.
// Param flattening assumed to follow jax pytree order (sorted dict keys):
//   layers[0..1]{ a_rel, kqv{item,tag,user}{b,w}, m_rel, out{item,tag,user}{b,w},
//                 p_rel, skip{item,tag,user} }, then proj{item,tag,user}{b,w}.
// Internal node-type order: user=0, item=1, tag=2.
// ============================================================================
extern "C" void kernel_launch(void* const* d_in, const int* in_sizes, int n_in,
                              void* d_out, int out_size, void* d_ws, size_t ws_size,
                              hipStream_t stream) {
  (void)out_size; (void)ws_size;
  if (n_in < 53) return;
  auto F = [&](int i) { return (const float*)d_in[i]; };
  auto I = [&](int i) { return (const int*)d_in[i]; };

  const int FD[3] = {128, 256, 64};
  const float* xraw[3] = {F(0), F(1), F(2)};
  int Nn[3];
  for (int t = 0; t < 3; ++t) Nn[t] = in_sizes[t] / FD[t];
  const int rowOff[3] = {0, Nn[0], Nn[0] + Nn[1]};
  const int TOT = Nn[0] + Nn[1] + Nn[2];

  struct Lin { const float* w; const float* b; };
  struct Layer {
    const float* a_rel; Lin kqv[3]; const float* m_rel; Lin out[3];
    const float* p_rel; const float* skip[3];
  };
  Layer L[2]; Lin proj[3];
  {
    const int ord[3] = {1, 2, 0};  // sorted keys: item, tag, user
    int i = 11;
    for (int l = 0; l < 2; ++l) {
      L[l].a_rel = F(i++);
      for (int j = 0; j < 3; ++j) { int t = ord[j]; L[l].kqv[t].b = F(i++); L[l].kqv[t].w = F(i++); }
      L[l].m_rel = F(i++);
      for (int j = 0; j < 3; ++j) { int t = ord[j]; L[l].out[t].b = F(i++); L[l].out[t].w = F(i++); }
      L[l].p_rel = F(i++);
      for (int j = 0; j < 3; ++j) { int t = ord[j]; L[l].skip[t] = F(i++); }
    }
    for (int j = 0; j < 3; ++j) { int t = ord[j]; proj[t].b = F(i++); proj[t].w = F(i++); }
  }

  // edge types: (src,dst): user->item, item->user, item->tag, tag->item
  struct Edge { int s, t, E; const int *si, *di; size_t cOff, aOff; } Eg[4];
  const int esrc[4] = {0, 1, 1, 2}, edst[4] = {1, 0, 2, 1};
  size_t cOff = 0, aOff = 0;
  for (int e = 0; e < 4; ++e) {
    Eg[e] = {esrc[e], edst[e], in_sizes[3 + 2 * e], I(3 + 2 * e), I(4 + 2 * e), cOff, aOff};
    cOff += (size_t)Nn[esrc[e]];
    aOff += (size_t)Eg[e].E;
  }
  const size_t totalE = aOff, relRows = cOff;

  // workspace layout (floats, regions 256B-aligned)
  float* Wp = (float*)d_ws;
  size_t o = 0;
  auto wsa = [&](size_t n) { size_t r = o; o += (n + 63) & ~(size_t)63; return r; };
  float*    X   = Wp + wsa((size_t)TOT * 128);
  float*    KQV = Wp + wsa((size_t)TOT * 384);
  float*    KR  = Wp + wsa(relRows * 128);
  float*    VR  = Wp + wsa(relRows * 128);
  float*    AB  = Wp + wsa(totalE * 4);
  unsigned* AM  = (unsigned*)(Wp + wsa((size_t)TOT * 4));
  float*    DN  = Wp + wsa((size_t)TOT * 4);
  float*    AG  = Wp + wsa((size_t)TOT * 128);

  // Stage 1: x = relu(x_raw @ Wproj + b)
  for (int t = 0; t < 3; ++t)
    hgt_gemm<E_RELU><<<dim3((Nn[t] + 31) / 32, 2), 256, 0, stream>>>(
        xraw[t], proj[t].w, proj[t].b, X + (size_t)rowOff[t] * 128,
        Nn[t], FD[t], 128, nullptr, nullptr);

  for (int l = 0; l < 2; ++l) {
    // fused kqv: [N,128] @ [128,384] + b   (cols: K | Q | V)
    for (int t = 0; t < 3; ++t)
      hgt_gemm<E_NONE><<<dim3((Nn[t] + 31) / 32, 6), 256, 0, stream>>>(
          X + (size_t)rowOff[t] * 128, L[l].kqv[t].w, L[l].kqv[t].b,
          KQV + (size_t)rowOff[t] * 384, Nn[t], 128, 384, nullptr, nullptr);

    // per-edge-type K/V relation transforms (per-node, per-head 32x32 WMMA)
    for (int e = 0; e < 4; ++e) {
      int s = Eg[e].s, Ns = Nn[s];
      dim3 g((Ns + 31) / 32, 4);
      hgt_rel<<<g, 128, 0, stream>>>(KQV + (size_t)rowOff[s] * 384,
                                     L[l].a_rel + e * 4096, KR + Eg[e].cOff * 128, Ns);
      hgt_rel<<<g, 128, 0, stream>>>(KQV + (size_t)rowOff[s] * 384 + 256,
                                     L[l].m_rel + e * 4096, VR + Eg[e].cOff * 128, Ns);
    }

    // init softmax state: amax = map(-inf), den = 0, agg = 0
    hgt_fill_u32<<<((size_t)TOT * 4 + 255) / 256, 256, 0, stream>>>(AM, 0x007FFFFFu, (size_t)TOT * 4);
    hgt_fill_f32<<<((size_t)TOT * 4 + 255) / 256, 256, 0, stream>>>(DN, 0.f, (size_t)TOT * 4);
    hgt_fill_f32<<<((size_t)TOT * 128 + 255) / 256, 256, 0, stream>>>(AG, 0.f, (size_t)TOT * 128);

    for (int e = 0; e < 4; ++e)
      hgt_pass1<<<(Eg[e].E + 7) / 8, 256, 0, stream>>>(
          KR + Eg[e].cOff * 128, KQV + (size_t)rowOff[Eg[e].t] * 384 + 128,
          Eg[e].si, Eg[e].di, Eg[e].E, L[l].p_rel + e * 4,
          AB + Eg[e].aOff * 4, AM, rowOff[Eg[e].t]);
    for (int e = 0; e < 4; ++e)
      hgt_pass2<<<((size_t)Eg[e].E * 4 + 255) / 256, 256, 0, stream>>>(
          AB + Eg[e].aOff * 4, Eg[e].di, Eg[e].E * 4, AM, DN, rowOff[Eg[e].t]);
    for (int e = 0; e < 4; ++e)
      hgt_pass3<<<(Eg[e].E + 7) / 8, 256, 0, stream>>>(
          AB + Eg[e].aOff * 4, VR + Eg[e].cOff * 128, Eg[e].si, Eg[e].di,
          Eg[e].E, DN, AG, rowOff[Eg[e].t]);

    hgt_gelu<<<((size_t)TOT * 128 + 255) / 256, 256, 0, stream>>>(AG, (size_t)TOT * 128);

    // out linear + gated skip + layer relu. Layer 0 in-place on X is safe:
    // xprev is read and rewritten at the same (row,col) by the same lane.
    float* xout = (l == 1) ? (float*)d_out : X;
    for (int t = 0; t < 3; ++t)
      hgt_gemm<E_OUT><<<dim3((Nn[t] + 31) / 32, 2), 256, 0, stream>>>(
          AG + (size_t)rowOff[t] * 128, L[l].out[t].w, L[l].out[t].b,
          xout + (size_t)rowOff[t] * 128, Nn[t], 128, 128,
          X + (size_t)rowOff[t] * 128, L[l].skip[t]);
  }
}
